// split_linear_44220983280023
// MI455X (gfx1250) — matrix operand order repaired and checked by reading, not compiled, not run on hardware
//
#include <hip/hip_runtime.h>

// out[b,i,o] = sum_f x[b,i,f] * W[i,o,f]
// 64 independent GEMMs (per i): X_i(256x1024) * W_i^T(1024x1024).
// Strategy: fp32 -> (hi,lo) bf16 split, 3x v_wmma_f32_16x16x32_bf16 per tile
// (hi*hi + hi*lo + lo*hi) with fp32 accumulation => near-fp32 accuracy at
// bf16 matrix-core rates; kernel sits at the ~17us HBM roofline (403MB @ 23.3TB/s).

#define IN_DIM 64
#define FEAT   1024
#define BATCH  256

#define BM 128
#define BN 128
#define BK 32
#define NK (FEAT / BK)   // 32 k-steps
#define LDP 40           // LDS row pitch in bf16 elems: 80B = 16B-aligned, bank-rotating

typedef __attribute__((ext_vector_type(16))) __bf16 v16bf;
typedef __attribute__((ext_vector_type(8)))  __bf16 v8bf;
typedef __attribute__((ext_vector_type(8)))  float  v8f;

__device__ __forceinline__ v16bf join8(v8bf a, v8bf b) {
  return __builtin_shufflevector(a, b, 0,1,2,3,4,5,6,7,8,9,10,11,12,13,14,15);
}

// Split 16 fp32 (4x float4) into hi/lo bf16 planes and store 2x16B per plane.
__device__ __forceinline__ void split_store16(__bf16* __restrict__ hiP,
                                              __bf16* __restrict__ loP,
                                              const float4* q) {
  float v[16];
#pragma unroll
  for (int j = 0; j < 4; ++j) {
    v[j*4+0] = q[j].x; v[j*4+1] = q[j].y; v[j*4+2] = q[j].z; v[j*4+3] = q[j].w;
  }
  v8bf h0, h1, l0, l1;
#pragma unroll
  for (int e = 0; e < 8; ++e) {
    __bf16 h = (__bf16)v[e];
    h0[e] = h; l0[e] = (__bf16)(v[e] - (float)h);
  }
#pragma unroll
  for (int e = 0; e < 8; ++e) {
    __bf16 h = (__bf16)v[8 + e];
    h1[e] = h; l1[e] = (__bf16)(v[8 + e] - (float)h);
  }
  *(v8bf*)(hiP)     = h0;
  *(v8bf*)(hiP + 8) = h1;
  *(v8bf*)(loP)     = l0;
  *(v8bf*)(loP + 8) = l1;
}

__global__ __launch_bounds__(256)
void split_linear_bf16x3_wmma(const float* __restrict__ x,
                              const float* __restrict__ W,
                              float* __restrict__ out) {
  __shared__ __bf16 Ahi[BM][LDP];
  __shared__ __bf16 Alo[BM][LDP];
  __shared__ __bf16 Bhi[BN][LDP];
  __shared__ __bf16 Blo[BN][LDP];

  const int o0 = blockIdx.x * BN;       // output-feature tile
  const int b0 = blockIdx.y * BM;       // batch tile
  const int i  = blockIdx.z;            // independent GEMM slice

  const int t    = threadIdx.x;
  const int srow = t >> 1;              // staging row 0..127
  const int scol = (t & 1) << 4;        // staging col 0 or 16 (of 32)

  // x[b][i][f]: b-stride = IN_DIM*FEAT, contiguous f.  W[i][o][f]: contiguous f.
  const float* xg = x + (size_t)(b0 + srow) * (IN_DIM * FEAT) + (size_t)i * FEAT + scol;
  const float* wg = W + (size_t)i * FEAT * FEAT + (size_t)(o0 + srow) * FEAT + scol;

  // wave mapping: 8 waves in 4(M) x 2(N); wave tile = 32x64 => 2x4 16x16 C frags
  const int wave = t >> 5;
  const int lane = t & 31;
  const int hl   = lane >> 4;           // lane half (0: lanes 0-15, 1: lanes 16-31)
  const int lm   = lane & 15;
  const int wrow = (wave & 3) * 32;
  const int wcol = (wave >> 2) * 64;

  v8f acc[2][4];
#pragma unroll
  for (int mf = 0; mf < 2; ++mf)
#pragma unroll
    for (int nf = 0; nf < 4; ++nf)
      acc[mf][nf] = (v8f){0.f, 0.f, 0.f, 0.f, 0.f, 0.f, 0.f, 0.f};

  float4 curA[4], curB[4], nxtA[4], nxtB[4];
#pragma unroll
  for (int j = 0; j < 4; ++j) {
    curA[j] = ((const float4*)xg)[j];
    curB[j] = ((const float4*)wg)[j];
  }

  for (int kk = 0; kk < NK; ++kk) {
    // ---- stage current K-slab into LDS as split hi/lo bf16 ----
    split_store16(&Ahi[srow][scol], &Alo[srow][scol], curA);
    split_store16(&Bhi[srow][scol], &Blo[srow][scol], curB);
    __syncthreads();

    // ---- issue next K-slab's global loads early (overlap with WMMA) ----
    if (kk + 1 < NK) {
      const float* xn = xg + (size_t)(kk + 1) * BK;
      const float* wn = wg + (size_t)(kk + 1) * BK;
#pragma unroll
      for (int j = 0; j < 4; ++j) {
        nxtA[j] = ((const float4*)xn)[j];
        nxtB[j] = ((const float4*)wn)[j];
      }
      if (kk + 2 < NK) {
        __builtin_prefetch(xg + (size_t)(kk + 2) * BK, 0, 1);  // global_prefetch_b8
        __builtin_prefetch(wg + (size_t)(kk + 2) * BK, 0, 1);
      }
    }

    // ---- load WMMA fragments (ISA 7.12.2 layouts) ----
    // A 16x32 bf16: lane m -> row m, K{0-7,16-23}; lane m+16 -> K{8-15,24-31}
    v16bf ah[2], al[2], bh[4], bl[4];
#pragma unroll
    for (int mf = 0; mf < 2; ++mf) {
      const int r = wrow + mf * 16 + lm;
      ah[mf] = join8(*(const v8bf*)&Ahi[r][hl * 8], *(const v8bf*)&Ahi[r][16 + hl * 8]);
      al[mf] = join8(*(const v8bf*)&Alo[r][hl * 8], *(const v8bf*)&Alo[r][16 + hl * 8]);
    }
    // B 32x16 bf16: lane n -> column n K{0-7,16-23}; lane n+16 -> K{8-15,24-31}
#pragma unroll
    for (int nf = 0; nf < 4; ++nf) {
      const int r = wcol + nf * 16 + lm;
      bh[nf] = join8(*(const v8bf*)&Bhi[r][hl * 8], *(const v8bf*)&Bhi[r][16 + hl * 8]);
      bl[nf] = join8(*(const v8bf*)&Blo[r][hl * 8], *(const v8bf*)&Blo[r][16 + hl * 8]);
    }

    // ---- bf16x3 matrix-core math: C += Ah*Bh + Ah*Bl + Al*Bh ----
#pragma unroll
    for (int mf = 0; mf < 2; ++mf)
#pragma unroll
      for (int nf = 0; nf < 4; ++nf) {
        v8f c = acc[mf][nf];
        c = __builtin_amdgcn_wmma_f32_16x16x32_bf16(false, ah[mf], false, bh[nf],
                                                    (short)0, c, false, false);
        c = __builtin_amdgcn_wmma_f32_16x16x32_bf16(false, ah[mf], false, bl[nf],
                                                    (short)0, c, false, false);
        c = __builtin_amdgcn_wmma_f32_16x16x32_bf16(false, al[mf], false, bh[nf],
                                                    (short)0, c, false, false);
        acc[mf][nf] = c;
      }

    __syncthreads();   // all reads done before next stage overwrites LDS
#pragma unroll
    for (int j = 0; j < 4; ++j) { curA[j] = nxtA[j]; curB[j] = nxtB[j]; }
  }

  // ---- epilogue: C/D layout -> VGPR r: lanes 0-15 row r, lanes 16-31 row r+8 ----
#pragma unroll
  for (int mf = 0; mf < 2; ++mf) {
#pragma unroll
    for (int nf = 0; nf < 4; ++nf) {
      const int mbase = b0 + wrow + mf * 16 + hl * 8;
      const int n     = o0 + wcol + nf * 16 + lm;
#pragma unroll
      for (int r = 0; r < 8; ++r) {
        out[(size_t)(mbase + r) * (IN_DIM * FEAT) + (size_t)i * FEAT + n] = acc[mf][nf][r];
      }
    }
  }
}

extern "C" void kernel_launch(void* const* d_in, const int* in_sizes, int n_in,
                              void* d_out, int out_size, void* d_ws, size_t ws_size,
                              hipStream_t stream) {
  (void)in_sizes; (void)n_in; (void)out_size; (void)d_ws; (void)ws_size;
  const float* x = (const float*)d_in[0];  // [256][64][1024] fp32
  const float* W = (const float*)d_in[1];  // [64][1024][1024] fp32
  float* out = (float*)d_out;              // [256][64][1024] fp32

  dim3 grid(FEAT / BN, BATCH / BM, IN_DIM);  // (8, 2, 64)
  dim3 block(256);                           // 8 wave32s
  split_linear_bf16x3_wmma<<<grid, block, 0, stream>>>(x, W, out);
}
